// OskarLayer_35124242546986
// MI455X (gfx1250) — matrix operand order repaired and checked by reading, not compiled, not run on hardware
//
#include <hip/hip_runtime.h>
#include <hip/hip_bf16.h>

// ---------------------------------------------------------------------------
// MI455X (gfx1250) linear-attention transformer layer.
// All heavy contractions run through v_wmma_f32_16x16x32_bf16 (wave32 WMMA),
// fp32 accumulation, fused epilogues. Global->LDS staging uses the CDNA5
// async-to-LDS path (GLOBAL_LOAD_ASYNC_TO_LDS_B128, ASYNCcnt) instead of the
// VGPR round trip: ~420 GFLOP vs <1 GB HBM traffic -> matrix-pipe bound.
// ---------------------------------------------------------------------------

typedef __bf16 bf16;
typedef __attribute__((ext_vector_type(16))) __bf16 bf16x16;
typedef __attribute__((ext_vector_type(8)))  float  f32x8;

#define HIDDEN 1024
#define NHEAD  16
#define HDIM   64
#define NSEQ   4096
#define NB     4
#define INTERD 4096

#define BM 128
#define BN 128
#define BK 32
#define LDT 40   // padded LDS row length (bf16 elems): 80B rows spread banks

enum { EPI_Q = 0, EPI_KT, EPI_VT, EPI_BF16, EPI_CTXOUT, EPI_F32BIAS, EPI_GELU };

__device__ __forceinline__ float elu1f(float x) {
  return x > 0.f ? x + 1.f : __expf(x);       // elu(x)+1
}
__device__ __forceinline__ float geluf(float x) {
  const float c = 0.7978845608028654f;        // sqrt(2/pi), tanh approx (jax default)
  float t = tanhf(c * (x + 0.044715f * x * x * x));
  return 0.5f * x * (1.f + t);
}

__device__ __forceinline__ unsigned lds_lo32(const void* p) {
  // flat LDS pointer -> LDS_ADDR.U32 (ISA: LDS addr = addr[31:0])
  return (unsigned)(unsigned long long)p;
}

// Issue one 32-byte row-slab (two b128) global -> LDS async copy for this
// thread. Uses ASYNCcnt; the same inst offset applies to both the LDS and
// global addresses (ISA 08_async_tensor §4.4), so one base each suffices.
__device__ __forceinline__ void async_tile(const bf16* __restrict__ base, int rows,
                                           int K, int tile0, int kt, int ldr,
                                           int s16, bf16* sm) {
  const int gr = tile0 + ldr;
  if (gr < rows) {
    const unsigned lds = lds_lo32(&sm[ldr * LDT + s16]);
    const bf16* g = base + (long)gr * K + (long)kt * BK + s16;
    asm volatile(
        "global_load_async_to_lds_b128 %0, %1, off\n\t"
        "global_load_async_to_lds_b128 %0, %1, off offset:16"
        :
        : "v"(lds), "v"(g)
        : "memory");
  }
}
__device__ __forceinline__ void wait_async0() {
  asm volatile("s_wait_asynccnt 0" ::: "memory");
}

// C[M,N] = A[M,K] @ BT[N,K]^T  (+bias, + fused epilogue). A, BT bf16 row-major.
template <int EPI>
__launch_bounds__(256, 2)
__global__ void gemm_bf16_wmma(const bf16* __restrict__ A, long sAb,
                               const bf16* __restrict__ BT, long sBb,
                               const float* __restrict__ bias,
                               float* __restrict__ outF,
                               bf16* __restrict__ outB,
                               int M, int N, int K,
                               const float* __restrict__ aux) {
  __shared__ __align__(16) bf16 smA[2][BM * LDT];
  __shared__ __align__(16) bf16 smB[2][BN * LDT];

  const int tid   = threadIdx.x;
  const int lane  = tid & 31;
  const int wave  = tid >> 5;
  const int wm    = wave & 3;    // 4 row-waves  -> 128 rows
  const int wn    = wave >> 2;   // 2 col-waves  -> 128 cols
  const int bm0   = blockIdx.y * BM;
  const int bn0   = blockIdx.x * BN;
  const int batch = blockIdx.z;
  A  += (long)batch * sAb;
  BT += (long)batch * sBb;

  const int l16   = lane & 15;
  const int koffA = (lane & 16) ? 8  : 0;   // A frag: K pairs  (16-bit A 16x32 layout)
  const int koffB = (lane & 16) ? 8  : 0;   // B frag: K pairs  (matches A frag k order)

  const int ldr = tid >> 1;        // tile row loaded by this thread (0..127)
  const int s16 = (tid & 1) * 16;  // element offset within 32-wide K slab

  // Partial tiles (the 64x64 head GEMMs): pre-zero LDS once; async loads only
  // overwrite in-range rows, OOB rows contribute exact zeros to the WMMAs.
  if ((M & (BM - 1)) || (N & (BN - 1))) {
    unsigned* za = (unsigned*)&smA[0][0];
    unsigned* zb = (unsigned*)&smB[0][0];
    for (int i = tid; i < (2 * BM * LDT) / 2; i += 256) {
      za[i] = 0u;
      zb[i] = 0u;
    }
    __syncthreads();
  }

  f32x8 acc[2][4];
  const f32x8 zero = {0.f, 0.f, 0.f, 0.f, 0.f, 0.f, 0.f, 0.f};
#pragma unroll
  for (int i = 0; i < 2; ++i)
#pragma unroll
    for (int j = 0; j < 4; ++j) acc[i][j] = zero;

  const int nk = K / BK;

  async_tile(A, M, K, bm0, 0, ldr, s16, smA[0]);
  async_tile(BT, N, K, bn0, 0, ldr, s16, smB[0]);
  wait_async0();
  __syncthreads();

  for (int kt = 0; kt < nk; ++kt) {
    const int cur = kt & 1;
    const bool hasNext = (kt + 1) < nk;
    if (hasNext) {
      async_tile(A, M, K, bm0, kt + 1, ldr, s16, smA[cur ^ 1]);
      async_tile(BT, N, K, bn0, kt + 1, ldr, s16, smB[cur ^ 1]);
      if (kt + 2 < nk) {  // global_prefetch_b8 for the slab after next
        __builtin_prefetch(A + (long)(bm0 + ldr) * K + (long)(kt + 2) * BK, 0, 0);
        __builtin_prefetch(BT + (long)(bn0 + ldr) * K + (long)(kt + 2) * BK, 0, 0);
      }
    }

    // fragments from LDS (two ds_load_b128 per fragment)
    bf16x16 afr[2], bfr[4];
#pragma unroll
    for (int i = 0; i < 2; ++i) {
      const int row = wm * 32 + i * 16 + l16;
      union { bf16x16 v; uint4 u[2]; } t;
      t.u[0] = *(const uint4*)&smA[cur][row * LDT + koffA];
      t.u[1] = *(const uint4*)&smA[cur][row * LDT + 16 + koffA];
      afr[i] = t.v;
    }
#pragma unroll
    for (int j = 0; j < 4; ++j) {
      const int col = wn * 64 + j * 16 + l16;
      union { bf16x16 v; uint4 u[2]; } t;
      t.u[0] = *(const uint4*)&smB[cur][col * LDT + koffB];
      t.u[1] = *(const uint4*)&smB[cur][col * LDT + 16 + koffB];
      bfr[j] = t.v;
    }
#pragma unroll
    for (int i = 0; i < 2; ++i)
#pragma unroll
      for (int j = 0; j < 4; ++j)
        acc[i][j] = __builtin_amdgcn_wmma_f32_16x16x32_bf16(
            false, afr[i], false, bfr[j], (short)0, acc[i][j], false, false);

    if (hasNext) wait_async0();  // next-buffer async copies have landed
    __syncthreads();
  }

  // --------------------------- fused epilogues -----------------------------
#pragma unroll
  for (int i = 0; i < 2; ++i) {
#pragma unroll
    for (int j = 0; j < 4; ++j) {
      const int gmb = bm0 + wm * 32 + i * 16 + ((lane >> 4) * 8);
      const int gn  = bn0 + wn * 64 + j * 16 + l16;
      if (gn >= N) continue;
#pragma unroll
      for (int r = 0; r < 8; ++r) {
        const int m = gmb + r;
        if (m >= M) continue;
        float v = acc[i][j][r];
        if constexpr (EPI == EPI_Q) {
          // phi(q) = elu+1, scatter to (B,H,N,d) bf16
          v = elu1f(v + bias[gn]);
          const int b = m / NSEQ, n = m % NSEQ, h = gn >> 6, e = gn & 63;
          outB[((((long)b * NHEAD + h) * NSEQ + n) << 6) + e] = (bf16)v;
        } else if constexpr (EPI == EPI_KT) {
          // phi(k)*mask, scatter transposed to (B,H,d,N) bf16
          v = elu1f(v + bias[gn]) * aux[m];
          const int b = m / NSEQ, n = m % NSEQ, h = gn >> 6, e = gn & 63;
          outB[(((long)(b * NHEAD + h) * HDIM + e) * NSEQ) + n] = (bf16)v;
        } else if constexpr (EPI == EPI_VT) {
          v = v + bias[gn];
          const int b = m / NSEQ, n = m % NSEQ, h = gn >> 6, e = gn & 63;
          outB[(((long)(b * NHEAD + h) * HDIM + e) * NSEQ) + n] = (bf16)v;
        } else if constexpr (EPI == EPI_BF16) {
          outB[(long)batch * M * N + (long)m * N + gn] = (bf16)v;
        } else if constexpr (EPI == EPI_CTXOUT) {
          // context row scaled by D^-1, scatter to (B,N,HID) bf16
          v *= aux[(long)batch * M + m];
          const int b = batch >> 4, h = batch & 15;
          outB[((long)b * NSEQ + m) * HIDDEN + h * HDIM + gn] = (bf16)v;
        } else if constexpr (EPI == EPI_F32BIAS) {
          outF[(long)m * N + gn] = v + bias[gn];
        } else if constexpr (EPI == EPI_GELU) {
          outB[(long)m * N + gn] = (bf16)geluf(v + bias[gn]);
        }
      }
    }
  }
}

// ksum[bh*64+d] = sum_n kT[bh,d,n]
__global__ void ksum_kernel(const bf16* __restrict__ kT, float* __restrict__ ksum) {
  __shared__ float red[256];
  const long row = blockIdx.x;
  const bf16* p = kT + row * NSEQ;
  float s = 0.f;
  for (int i = threadIdx.x; i < NSEQ; i += 256) s += (float)p[i];
  red[threadIdx.x] = s;
  __syncthreads();
  for (int st = 128; st > 0; st >>= 1) {
    if (threadIdx.x < st) red[threadIdx.x] += red[threadIdx.x + st];
    __syncthreads();
  }
  if (threadIdx.x == 0) ksum[row] = red[0];
}

// dinv[bh,n] = 1 / dot(q[bh,n,:], ksum[bh,:])
__global__ void dinv_kernel(const bf16* __restrict__ q,
                            const float* __restrict__ ksum,
                            float* __restrict__ dinv) {
  const long idx = (long)blockIdx.x * 256 + threadIdx.x;  // bh*NSEQ + n
  const long bh  = idx >> 12;                             // /NSEQ
  const bf16* qp = q + idx * HDIM;
  const float* ks = ksum + bh * HDIM;
  float s = 0.f;
#pragma unroll 8
  for (int e = 0; e < HDIM; ++e) s += (float)qp[e] * ks[e];
  dinv[idx] = 1.f / s;
}

// out = LN(x + y)*g + b ; optional bf16 mirror. Safe when outF aliases x or y.
__global__ void add_ln_kernel(const float* __restrict__ x,
                              const float* __restrict__ y,
                              const float* __restrict__ g,
                              const float* __restrict__ b,
                              float* __restrict__ outF,
                              bf16* __restrict__ outB) {
  __shared__ float red[256], red2[256];
  const long row = blockIdx.x;
  const float* xp = x + row * HIDDEN;
  const float* yp = y + row * HIDDEN;
  float v[4], s = 0.f, s2 = 0.f;
#pragma unroll
  for (int t = 0; t < 4; ++t) {
    const int c = threadIdx.x + t * 256;
    v[t] = xp[c] + yp[c];
    s += v[t];
    s2 += v[t] * v[t];
  }
  red[threadIdx.x] = s;
  red2[threadIdx.x] = s2;
  __syncthreads();
  for (int st = 128; st > 0; st >>= 1) {
    if (threadIdx.x < st) {
      red[threadIdx.x] += red[threadIdx.x + st];
      red2[threadIdx.x] += red2[threadIdx.x + st];
    }
    __syncthreads();
  }
  const float mean = red[0] * (1.f / HIDDEN);
  const float var  = red2[0] * (1.f / HIDDEN) - mean * mean;
  const float rstd = rsqrtf(var + 1e-12f);
  __syncthreads();
#pragma unroll
  for (int t = 0; t < 4; ++t) {
    const int c = threadIdx.x + t * 256;
    const float o = (v[t] - mean) * rstd * g[c] + b[c];
    outF[row * HIDDEN + c] = o;
    if (outB) outB[row * HIDDEN + c] = (bf16)o;
  }
}

// f32 (R x C) -> bf16 (C x R)   (weights pre-transposed so GEMM B-tiles load
// row-contiguous and WMMA B fragments are two b128 LDS reads)
__global__ void transpose_to_bf16(const float* __restrict__ in,
                                  bf16* __restrict__ out, int R, int C) {
  __shared__ float t[32][33];
  const int tx = threadIdx.x & 31, ty = threadIdx.x >> 5;
  const int c0 = blockIdx.x * 32, r0 = blockIdx.y * 32;
#pragma unroll
  for (int i = 0; i < 4; ++i)
    t[ty + i * 8][tx] = in[(long)(r0 + ty + i * 8) * C + c0 + tx];
  __syncthreads();
#pragma unroll
  for (int i = 0; i < 4; ++i) {
    const int rr = ty + i * 8;
    out[(long)(c0 + rr) * R + r0 + tx] = (bf16)t[tx][rr];
  }
}

__global__ void f32_to_bf16(const float* __restrict__ in, bf16* __restrict__ out,
                            long n) {
  const long i = ((long)blockIdx.x * 256 + threadIdx.x) * 4;
  if (i + 3 < n) {
    const float4 f = *(const float4*)(in + i);
    out[i + 0] = (bf16)f.x;
    out[i + 1] = (bf16)f.y;
    out[i + 2] = (bf16)f.z;
    out[i + 3] = (bf16)f.w;
  }
}

extern "C" void kernel_launch(void* const* d_in, const int* in_sizes, int n_in,
                              void* d_out, int out_size, void* d_ws, size_t ws_size,
                              hipStream_t stream) {
  (void)in_sizes; (void)n_in; (void)out_size; (void)ws_size;
  const float* x    = (const float*)d_in[0];
  const float* mask = (const float*)d_in[1];
  const float* wq   = (const float*)d_in[2];
  const float* bq   = (const float*)d_in[3];
  const float* wk   = (const float*)d_in[4];
  const float* bk   = (const float*)d_in[5];
  const float* wv   = (const float*)d_in[6];
  const float* bv   = (const float*)d_in[7];
  const float* wd   = (const float*)d_in[8];
  const float* bd   = (const float*)d_in[9];
  const float* gla  = (const float*)d_in[10];
  const float* bla  = (const float*)d_in[11];
  const float* wf   = (const float*)d_in[12];
  const float* bff  = (const float*)d_in[13];
  const float* wfo  = (const float*)d_in[14];
  const float* bfo  = (const float*)d_in[15];
  const float* glf  = (const float*)d_in[16];
  const float* blf  = (const float*)d_in[17];
  float* out = (float*)d_out;

  const long MTOK = (long)NB * NSEQ;  // 16384 rows
  char* p = (char*)d_ws;
  auto alloc = [&](size_t bytes) -> char* {
    char* r = p;
    p += (bytes + 255) & ~(size_t)255;
    return r;
  };
  bf16* xb    = (bf16*)alloc(MTOK * HIDDEN * 2);
  bf16* wqT   = (bf16*)alloc((long)HIDDEN * HIDDEN * 2);
  bf16* wkT   = (bf16*)alloc((long)HIDDEN * HIDDEN * 2);
  bf16* wvT   = (bf16*)alloc((long)HIDDEN * HIDDEN * 2);
  bf16* wdT   = (bf16*)alloc((long)HIDDEN * HIDDEN * 2);
  bf16* wfT   = (bf16*)alloc((long)HIDDEN * INTERD * 2);
  bf16* wfoT  = (bf16*)alloc((long)HIDDEN * INTERD * 2);
  bf16* qb    = (bf16*)alloc(MTOK * HIDDEN * 2);   // (B,H,N,d)
  bf16* kT    = (bf16*)alloc(MTOK * HIDDEN * 2);   // (B,H,d,N)
  bf16* vT    = (bf16*)alloc(MTOK * HIDDEN * 2);   // (B,H,d,N)
  bf16* ctxT  = (bf16*)alloc((long)NB * NHEAD * HDIM * HDIM * 2);
  float* ksum = (float*)alloc((long)NB * NHEAD * HDIM * 4);
  float* dinv = (float*)alloc((long)NB * NHEAD * NSEQ * 4);
  bf16* ctxo  = (bf16*)alloc(MTOK * HIDDEN * 2);   // (B,N,HID)
  float* proj = (float*)alloc(MTOK * HIDDEN * 4);  // reused as h2
  bf16* attnb = (bf16*)alloc(MTOK * HIDDEN * 2);
  bf16* h1    = (bf16*)alloc(MTOK * INTERD * 2);

  const dim3 blk(256);
  const int BHT = NB * NHEAD;  // 64 head batches

  // stage inputs to bf16 / pre-transposed weights
  f32_to_bf16<<<dim3((unsigned)((MTOK * HIDDEN / 4 + 255) / 256)), blk, 0, stream>>>(
      x, xb, MTOK * HIDDEN);
  transpose_to_bf16<<<dim3(HIDDEN / 32, HIDDEN / 32), blk, 0, stream>>>(wq, wqT, HIDDEN, HIDDEN);
  transpose_to_bf16<<<dim3(HIDDEN / 32, HIDDEN / 32), blk, 0, stream>>>(wk, wkT, HIDDEN, HIDDEN);
  transpose_to_bf16<<<dim3(HIDDEN / 32, HIDDEN / 32), blk, 0, stream>>>(wv, wvT, HIDDEN, HIDDEN);
  transpose_to_bf16<<<dim3(HIDDEN / 32, HIDDEN / 32), blk, 0, stream>>>(wd, wdT, HIDDEN, HIDDEN);
  transpose_to_bf16<<<dim3(INTERD / 32, HIDDEN / 32), blk, 0, stream>>>(wf, wfT, HIDDEN, INTERD);
  transpose_to_bf16<<<dim3(HIDDEN / 32, INTERD / 32), blk, 0, stream>>>(wfo, wfoT, INTERD, HIDDEN);

  const dim3 gQKV(HIDDEN / BN, MTOK / BM, 1);
  gemm_bf16_wmma<EPI_Q><<<gQKV, blk, 0, stream>>>(
      xb, 0, wqT, 0, bq, nullptr, qb, (int)MTOK, HIDDEN, HIDDEN, nullptr);
  gemm_bf16_wmma<EPI_KT><<<gQKV, blk, 0, stream>>>(
      xb, 0, wkT, 0, bk, nullptr, kT, (int)MTOK, HIDDEN, HIDDEN, mask);
  gemm_bf16_wmma<EPI_VT><<<gQKV, blk, 0, stream>>>(
      xb, 0, wvT, 0, bv, nullptr, vT, (int)MTOK, HIDDEN, HIDDEN, nullptr);

  ksum_kernel<<<dim3(BHT * HDIM), blk, 0, stream>>>(kT, ksum);
  dinv_kernel<<<dim3((BHT * NSEQ) / 256), blk, 0, stream>>>(qb, ksum, dinv);

  // ctxT[bh, e, d] = sum_n v[n,e] k[n,d]  (batched 64x64, K=4096)
  gemm_bf16_wmma<EPI_BF16><<<dim3(1, 1, BHT), blk, 0, stream>>>(
      vT, (long)HDIM * NSEQ, kT, (long)HDIM * NSEQ, nullptr, nullptr, ctxT,
      HDIM, HDIM, NSEQ, nullptr);

  // context[bh, n, e] = dinv * (q[n,:] . ctxT[e,:]) -> scatter (B,N,HID)
  gemm_bf16_wmma<EPI_CTXOUT><<<dim3(1, NSEQ / BM, BHT), blk, 0, stream>>>(
      qb, (long)NSEQ * HDIM, ctxT, (long)HDIM * HDIM, nullptr, nullptr, ctxo,
      NSEQ, HDIM, HDIM, dinv);

  // dense projection + residual layernorm
  gemm_bf16_wmma<EPI_F32BIAS><<<gQKV, blk, 0, stream>>>(
      ctxo, 0, wdT, 0, bd, proj, nullptr, (int)MTOK, HIDDEN, HIDDEN, nullptr);
  add_ln_kernel<<<dim3((unsigned)MTOK), blk, 0, stream>>>(x, proj, gla, bla, out, attnb);

  // FFN
  gemm_bf16_wmma<EPI_GELU><<<dim3(INTERD / BN, MTOK / BM, 1), blk, 0, stream>>>(
      attnb, 0, wfT, 0, bff, nullptr, h1, (int)MTOK, INTERD, HIDDEN, nullptr);
  gemm_bf16_wmma<EPI_F32BIAS><<<gQKV, blk, 0, stream>>>(
      h1, 0, wfoT, 0, bfo, proj, nullptr, (int)MTOK, HIDDEN, INTERD, nullptr);
  add_ln_kernel<<<dim3((unsigned)MTOK), blk, 0, stream>>>(proj, out, glf, blf, out, nullptr);
}
